// MoE_27015344291985
// MI455X (gfx1250) — matrix operand order repaired and checked by reading, not compiled, not run on hardware
//
#include <hip/hip_runtime.h>

// ---------------------------------------------------------------------------
// MoE of 9 CNNs (1 gate + 8 experts) on gfx1250, bf16 WMMA (V_WMMA_F32_16X16X32_BF16)
// LDS-staged implicit-GEMM convs (NHWC), LDS-staged fc1, branchless fragments,
// async global->LDS staging (GLOBAL_LOAD_ASYNC_TO_LDS_B128) where available.
// ---------------------------------------------------------------------------

typedef __attribute__((ext_vector_type(16))) __bf16 bf16x16;
typedef __attribute__((ext_vector_type(8)))  __bf16 bf16x8;
typedef __attribute__((ext_vector_type(8)))  float  v8f;

// exact pointee type the async builtin expects: gcc-style vector_size(16) int
typedef int v4i_gcc __attribute__((vector_size(16)));
typedef __attribute__((address_space(1))) v4i_gcc* as1_v4i;
typedef __attribute__((address_space(3))) v4i_gcc* as3_v4i;

#define NB    2048
#define PIX1  784     // 28*28
#define PIX2  196     // 14*14
#define C1    32
#define C2    64
#define FLATK 3136    // 64*7*7
#define NFC1  128

#if defined(__gfx1250__) && __has_builtin(__builtin_amdgcn_global_load_async_to_lds_b128) && __has_builtin(__builtin_amdgcn_s_wait_asynccnt)
#define HAVE_ASYNC_LDS 1
#endif

__device__ __forceinline__ v8f wmma_bf16(bf16x16 a, bf16x16 b, v8f c) {
  return __builtin_amdgcn_wmma_f32_16x16x32_bf16(false, a, false, b,
                                                 (short)0, c, false, false);
}
__device__ __forceinline__ v8f zerov() { v8f z = {0,0,0,0,0,0,0,0}; return z; }
__device__ __forceinline__ bf16x8 zerov8h() {
  bf16x8 z;
#pragma unroll
  for (int i = 0; i < 8; ++i) z[i] = (__bf16)0.f;
  return z;
}
__device__ __forceinline__ bf16x16 pack16(bf16x8 lo, bf16x8 hi) {
  bf16x16 r;
#pragma unroll
  for (int i = 0; i < 8; ++i) { r[i] = lo[i]; r[8 + i] = hi[i]; }
  return r;
}

// one 16-byte global -> LDS copy per thread; async path when available
__device__ __forceinline__ void copy_b128_to_lds(__bf16* lds_dst, const __bf16* gsrc) {
#ifdef HAVE_ASYNC_LDS
  __builtin_amdgcn_global_load_async_to_lds_b128(
      (as1_v4i)gsrc, (as3_v4i)lds_dst, 0, 0);
#else
  *(bf16x8*)lds_dst = *(const bf16x8*)gsrc;
#endif
}
__device__ __forceinline__ void async_fence() {
#ifdef HAVE_ASYNC_LDS
  __builtin_amdgcn_s_wait_asynccnt(0);
#endif
}

// ---------------------------------------------------------------------------
// weight prep: conv2 W2[64,32,5,5] fp32 -> wtap[25][64][32] bf16
// ---------------------------------------------------------------------------
__global__ __launch_bounds__(256) void prep_w2(
    const float* __restrict__ W2, __bf16* __restrict__ wtap) {
  int i = blockIdx.x * 256 + threadIdx.x;
  if (i >= 25 * C2 * C1) return;
  int t = i / (C2 * C1);
  int f = i - t * (C2 * C1);                     // f = n*32 + c
  wtap[i] = (__bf16)W2[(size_t)f * 25 + t];
}

// weight prep: fc1 Wf1[128,3136] fp32 -> bf16 (same layout)
__global__ __launch_bounds__(256) void prep_wf1(
    const float* __restrict__ Wf1, __bf16* __restrict__ out) {
  int i = blockIdx.x * 256 + threadIdx.x;
  if (i >= NFC1 * FLATK) return;
  out[i] = (__bf16)Wf1[i];
}

// ---------------------------------------------------------------------------
// conv1: x[b,1,28,28] fp32 -> relu(conv5x5 pad2) -> y1 NHWC [b][784][32] bf16
// one block per image; 34x32 padded image (rows 32..33 zeroed so out-of-tap
// A elements are finite; B rows >= 25 are zero, so they contribute nothing)
// ---------------------------------------------------------------------------
__global__ __launch_bounds__(256) void conv1_wmma(
    const float* __restrict__ x, const float* __restrict__ W1,
    const float* __restrict__ B1, __bf16* __restrict__ y1) {
  __shared__ __align__(16) __bf16 ldsX[34 * 32];   // padded image + 2 spare rows
  __shared__ __align__(16) __bf16 ldsW1[32 * 32];  // [n][t], taps padded to 32 (zero)
  const int img = blockIdx.x;
  const float* xs = x + (size_t)img * PIX1;

  for (int i = threadIdx.x; i < 1024; i += 256) {
    int hh = i >> 5, ww = i & 31;
    int h = hh - 2, w = ww - 2;
    bool inb = ((unsigned)h < 28u) & ((unsigned)w < 28u);
    float v = xs[inb ? h * 28 + w : 0];            // clamped addr, no branch
    ldsX[i] = (__bf16)(inb ? v : 0.f);
    int n = hh, t = ww;
    float wv = W1[(t < 25) ? n * 25 + t : 0];
    ldsW1[i] = (__bf16)((t < 25) ? wv : 0.f);
  }
  if (threadIdx.x < 64) ldsX[1024 + threadIdx.x] = (__bf16)0.f;  // spare rows
  __syncthreads();

  const int lane = threadIdx.x & 31;
  const int wave = threadIdx.x >> 5;
  const int half = lane >> 4;
  const int lan  = lane & 15;

  for (int mt = wave; mt < 49; mt += 8) {          // 49 M-tiles of 16 pixels
    int p = mt * 16 + lan;
    int h = p / 28, w = p - (p / 28) * 28;
    int abase = (h + 2) * 32 + (w + 2);
    bf16x16 a;
#pragma unroll
    for (int e = 0; e < 16; ++e) {
      const int tL = (e < 8) ? e : e + 8;          // tap for half==0 (all < 25)
      const int tH = tL + 8;                       // tap for half==1 (may be >= 25)
      const int offL = (tL / 5 - 2) * 32 + (tL % 5 - 2);   // compile-time
      const int offH = (tH / 5 - 2) * 32 + (tH % 5 - 2);   // compile-time
      a[e] = ldsX[abase + (half ? offH : offL)];
    }
    v8f acc[2];
    acc[0] = zerov(); acc[1] = zerov();
#pragma unroll
    for (int nt = 0; nt < 2; ++nt) {
      const __bf16* wl = &ldsW1[(nt * 16 + lan) * 32 + half * 8];
      bf16x16 b = pack16(*(const bf16x8*)wl, *(const bf16x8*)(wl + 16));
      acc[nt] = wmma_bf16(a, b, acc[nt]);
    }
#pragma unroll
    for (int nt = 0; nt < 2; ++nt) {
      int ch = nt * 16 + lan;
      float bias = B1[ch];
#pragma unroll
      for (int r = 0; r < 8; ++r) {
        int pr = mt * 16 + r + half * 8;
        y1[((size_t)img * PIX1 + pr) * C1 + ch] = (__bf16)fmaxf(acc[nt][r] + bias, 0.f);
      }
    }
  }
}

// ---------------------------------------------------------------------------
// pool1: NHWC [b][784][32] -> NHWC [b][196][32] bf16
// ---------------------------------------------------------------------------
__global__ __launch_bounds__(256) void pool1_kernel(
    const __bf16* __restrict__ in, __bf16* __restrict__ out) {
  int idx = blockIdx.x * 256 + threadIdx.x;
  if (idx >= NB * PIX2 * C1) return;
  int c   = idx & 31;
  int pix = idx >> 5;
  int b   = pix / PIX2;
  int hw  = pix - b * PIX2;
  int oh = hw / 14, ow = hw - (hw / 14) * 14;
  const __bf16* s = in + ((size_t)b * PIX1 + (2 * oh) * 28 + 2 * ow) * C1 + c;
  float m = fmaxf(fmaxf((float)s[0], (float)s[C1]),
                  fmaxf((float)s[28 * C1], (float)s[29 * C1]));
  out[idx] = (__bf16)m;
}

// ---------------------------------------------------------------------------
// conv2: NHWC [b][196][32] -> relu(conv5x5 pad2) -> y2 NHWC [b][196][64] bf16
// one block per image; padded 18x18x32 NHWC input in LDS; per-tap 64x32 bf16
// weight slice staged via async global->LDS; K-step = 32 channels per tap.
// ---------------------------------------------------------------------------
__global__ __launch_bounds__(256) void conv2_wmma(
    const __bf16* __restrict__ p1, const __bf16* __restrict__ wtap,
    const float* __restrict__ B2, __bf16* __restrict__ y2) {
  __shared__ __align__(16) __bf16 ldsP[18 * 18 * C1];  // padded NHWC input
  __shared__ __align__(16) __bf16 ldsW[C2 * C1];       // per-tap [n][c]
  const int img = blockIdx.x;
  const __bf16* src = p1 + (size_t)img * (PIX2 * C1);

  // stage padded input, 8-channel vector units (branchless, zero halo)
  for (int u = threadIdx.x; u < 18 * 18 * 4; u += 256) {
    int pix = u >> 2, cg = (u & 3) * 8;
    int hh = pix / 18, ww = pix - (pix / 18) * 18;
    int h = hh - 2, w = ww - 2;
    bool inb = ((unsigned)h < 14u) & ((unsigned)w < 14u);
    bf16x8 v = *(const bf16x8*)(src + (size_t)(inb ? h * 14 + w : 0) * C1 + cg);
    v = inb ? v : zerov8h();
    *(bf16x8*)&ldsP[(hh * 18 + ww) * C1 + cg] = v;
  }

  const int lane = threadIdx.x & 31;
  const int wave = threadIdx.x >> 5;
  const int half = lane >> 4;
  const int lan  = lane & 15;

  v8f acc[2][4];
#pragma unroll
  for (int u = 0; u < 2; ++u)
#pragma unroll
    for (int j = 0; j < 4; ++j) acc[u][j] = zerov();

  for (int t = 0; t < 25; ++t) {
    __syncthreads();
    // stage this tap's 64x32 weight slice: one async b128 per thread
    copy_b128_to_lds(&ldsW[threadIdx.x * 8],
                     wtap + (size_t)t * (C2 * C1) + threadIdx.x * 8);
    async_fence();
    __syncthreads();

    const int dy = t / 5 - 2, dx = t % 5 - 2;
#pragma unroll
    for (int u = 0; u < 2; ++u) {
      int mt = wave + u * 8;                       // wave-uniform guard
      if (mt < 13) {
        int p = mt * 16 + lan;
        p = (p < 196) ? p : 195;                   // clamp, stores guarded later
        int h = p / 14, w = p - (p / 14) * 14;
        const __bf16* ap = &ldsP[((h + dy + 2) * 18 + (w + dx + 2)) * C1 + half * 8];
        bf16x16 a = pack16(*(const bf16x8*)ap, *(const bf16x8*)(ap + 16));
#pragma unroll
        for (int j = 0; j < 4; ++j) {
          const __bf16* wl = &ldsW[(j * 16 + lan) * C1 + half * 8];
          bf16x16 b = pack16(*(const bf16x8*)wl, *(const bf16x8*)(wl + 16));
          acc[u][j] = wmma_bf16(a, b, acc[u][j]);
        }
      }
    }
  }

#pragma unroll
  for (int u = 0; u < 2; ++u) {
    int mt = wave + u * 8;
    if (mt < 13) {
#pragma unroll
      for (int j = 0; j < 4; ++j) {
        int ch = j * 16 + lan;
        float bias = B2[ch];
#pragma unroll
        for (int r = 0; r < 8; ++r) {
          int pr = mt * 16 + r + half * 8;
          if (pr < 196)
            y2[((size_t)img * PIX2 + pr) * C2 + ch] =
                (__bf16)fmaxf(acc[u][j][r] + bias, 0.f);
        }
      }
    }
  }
}

// ---------------------------------------------------------------------------
// pool2 + NCHW flatten: NHWC [b][196][64] -> [b][3136] bf16 (c*49+oh*7+ow)
// ---------------------------------------------------------------------------
__global__ __launch_bounds__(256) void pool2_kernel(
    const __bf16* __restrict__ in, __bf16* __restrict__ out) {
  int idx = blockIdx.x * 256 + threadIdx.x;
  if (idx >= NB * FLATK) return;
  int b  = idx / FLATK;
  int f  = idx - b * FLATK;
  int c  = f / 49;
  int r2 = f - c * 49;
  int oh = r2 / 7, ow = r2 - (r2 / 7) * 7;
  const __bf16* s = in + ((size_t)b * PIX2 + (2 * oh) * 14 + 2 * ow) * C2 + c;
  float m = fmaxf(fmaxf((float)s[0], (float)s[C2]),
                  fmaxf((float)s[14 * C2], (float)s[15 * C2]));
  out[idx] = (__bf16)m;
}

// ---------------------------------------------------------------------------
// fc1: relu(A[2048,3136] @ Wf1b^T + bf1) -> [2048,128] bf16
// block tile 64x128, A async-staged via LDS, B pre-converted bf16, 98 K-steps
// ---------------------------------------------------------------------------
__global__ __launch_bounds__(256) void fc1_wmma(
    const __bf16* __restrict__ A, const __bf16* __restrict__ Wf1b,
    const float* __restrict__ Bf1, __bf16* __restrict__ out) {
  __shared__ __align__(16) __bf16 ldsA[64 * 32];
  const int lane  = threadIdx.x & 31;
  const int wave  = threadIdx.x >> 5;
  const int half  = lane >> 4;
  const int lan   = lane & 15;
  const int mblk  = blockIdx.x * 64;
  const int mloc  = (wave >> 1) * 16;
  const int nbase = (wave & 1) * 64;

  v8f acc[4];
#pragma unroll
  for (int j = 0; j < 4; ++j) acc[j] = zerov();

  const int srow = threadIdx.x >> 2;
  const int skg  = (threadIdx.x & 3) * 8;
  for (int k0 = 0; k0 < FLATK; k0 += 32) {
    __syncthreads();
    copy_b128_to_lds(&ldsA[srow * 32 + skg],
                     A + (size_t)(mblk + srow) * FLATK + k0 + skg);
    async_fence();
    __syncthreads();

    const __bf16* ap = &ldsA[(mloc + lan) * 32 + half * 8];
    bf16x16 a = pack16(*(const bf16x8*)ap, *(const bf16x8*)(ap + 16));
#pragma unroll
    for (int j = 0; j < 4; ++j) {
      int n = nbase + j * 16 + lan;
      const __bf16* wp = Wf1b + (size_t)n * FLATK + k0 + half * 8;
      bf16x16 b = pack16(*(const bf16x8*)wp, *(const bf16x8*)(wp + 16));
      acc[j] = wmma_bf16(a, b, acc[j]);
    }
  }

#pragma unroll
  for (int j = 0; j < 4; ++j) {
    int col = nbase + j * 16 + lan;
    float bias = Bf1[col];
#pragma unroll
    for (int r = 0; r < 8; ++r) {
      int row = mblk + mloc + r + half * 8;
      out[(size_t)row * NFC1 + col] = (__bf16)fmaxf(acc[j][r] + bias, 0.f);
    }
  }
}

// ---------------------------------------------------------------------------
// fc2: A[2048,128] @ Wf2^T + bf2 -> raw logits [2048,16] fp32 (N padded)
// ---------------------------------------------------------------------------
__global__ __launch_bounds__(256) void fc2_wmma(
    const __bf16* __restrict__ A, const float* __restrict__ Wf2,
    const float* __restrict__ Bf2, int nvalid, float* __restrict__ logits) {
  const int lane = threadIdx.x & 31;
  const int wave = threadIdx.x >> 5;
  const int half = lane >> 4;
  const int lan  = lane & 15;
  const int m0   = blockIdx.x * 128 + wave * 16;

  const int   nc  = (lan < nvalid) ? lan : 0;      // clamp, no branch
  const float sel = (lan < nvalid) ? 1.f : 0.f;

  v8f acc = zerov();
#pragma unroll
  for (int k0 = 0; k0 < NFC1; k0 += 32) {
    const __bf16* ap = A + (size_t)(m0 + lan) * NFC1 + k0 + half * 16;
    bf16x16 a = pack16(*(const bf16x8*)ap, *(const bf16x8*)(ap + 8));
    const float* wp = Wf2 + (size_t)nc * NFC1 + k0 + half * 16;
    bf16x16 b;
#pragma unroll
    for (int i = 0; i < 16; ++i) b[i] = (__bf16)(wp[i] * sel);
    acc = wmma_bf16(a, b, acc);
  }

  float bias = (lan < nvalid) ? Bf2[nc] : 0.f;
#pragma unroll
  for (int r = 0; r < 8; ++r) {
    int row = m0 + r + half * 8;
    logits[(size_t)row * 16 + lan] = acc[r] + bias;
  }
}

// ---------------------------------------------------------------------------
// gating + combine
// ---------------------------------------------------------------------------
__global__ __launch_bounds__(256) void moe_combine(
    const float* __restrict__ logits, float* __restrict__ out) {
  int b = blockIdx.x * 256 + threadIdx.x;
  if (b >= NB) return;

  const float* g = logits + (size_t)b * 16;        // slot 0 = gate
  float p[8], mx = -1e30f;
#pragma unroll
  for (int e = 0; e < 8; ++e) mx = fmaxf(mx, g[e]);
  float s = 0.f;
#pragma unroll
  for (int e = 0; e < 8; ++e) { p[e] = __expf(g[e] - mx); s += p[e]; }
  float inv = 1.f / s;
#pragma unroll
  for (int e = 0; e < 8; ++e) p[e] *= inv;

  int ti[4]; float tv[4];
  bool used[8] = {false,false,false,false,false,false,false,false};
#pragma unroll
  for (int j = 0; j < 4; ++j) {
    int best = 0; float bv = -1e30f;
#pragma unroll
    for (int e = 0; e < 8; ++e)
      if (!used[e] && p[e] > bv) { bv = p[e]; best = e; }
    used[best] = true; ti[j] = best; tv[j] = bv;
  }
  float s2 = 0.f, gw[4];
#pragma unroll
  for (int j = 0; j < 4; ++j) { gw[j] = __expf(tv[j] - tv[0]); s2 += gw[j]; }
  float inv2 = 1.f / s2;

  float o[10] = {0,0,0,0,0,0,0,0,0,0};
  for (int j = 0; j < 4; ++j) {
    const float* el = logits + ((size_t)(ti[j] + 1) * NB + b) * 16;
    float m3 = -1e30f;
#pragma unroll
    for (int c = 0; c < 10; ++c) m3 = fmaxf(m3, el[c]);
    float qs[10], s3 = 0.f;
#pragma unroll
    for (int c = 0; c < 10; ++c) { qs[c] = __expf(el[c] - m3); s3 += qs[c]; }
    float wj = (gw[j] * inv2) / s3;
#pragma unroll
    for (int c = 0; c < 10; ++c) o[c] += wj * qs[c];
  }
#pragma unroll
  for (int c = 0; c < 10; ++c) out[(size_t)b * 10 + c] = o[c];
}

// ---------------------------------------------------------------------------
extern "C" void kernel_launch(void* const* d_in, const int* in_sizes, int n_in,
                              void* d_out, int out_size, void* d_ws, size_t ws_size,
                              hipStream_t stream) {
  const float* x    = (const float*)d_in[0];
  const float* gW1  = (const float*)d_in[1];
  const float* gb1  = (const float*)d_in[2];
  const float* gW2  = (const float*)d_in[3];
  const float* gb2  = (const float*)d_in[4];
  const float* gWf1 = (const float*)d_in[5];
  const float* gbf1 = (const float*)d_in[6];
  const float* gWf2 = (const float*)d_in[7];
  const float* gbf2 = (const float*)d_in[8];
  const float* eW1  = (const float*)d_in[9];
  const float* eb1  = (const float*)d_in[10];
  const float* eW2  = (const float*)d_in[11];
  const float* eb2  = (const float*)d_in[12];
  const float* eWf1 = (const float*)d_in[13];
  const float* ebf1 = (const float*)d_in[14];
  const float* eWf2 = (const float*)d_in[15];
  const float* ebf2 = (const float*)d_in[16];
  (void)in_sizes; (void)n_in; (void)out_size; (void)ws_size;

  char* ws = (char*)d_ws;
  size_t off = 0;
  auto alloc = [&](size_t bytes) {
    void* p = ws + off;
    off += (bytes + 255) & ~(size_t)255;
    return p;
  };
  __bf16* y1   = (__bf16*)alloc((size_t)NB * PIX1 * C1 * 2);   // conv1 out (NHWC)
  __bf16* pl1  = (__bf16*)alloc((size_t)NB * PIX2 * C1 * 2);   // pool1 out (NHWC)
  __bf16* y2   = (__bf16*)alloc((size_t)NB * PIX2 * C2 * 2);   // conv2 out (NHWC)
  __bf16* fl2  = (__bf16*)alloc((size_t)NB * FLATK * 2);       // pool2 out (NCHW flat)
  __bf16* f1o  = (__bf16*)alloc((size_t)NB * NFC1 * 2);        // fc1 out
  float*  lg   = (float*)alloc((size_t)9 * NB * 16 * 4);       // per-slot logits
  __bf16* wtap = (__bf16*)alloc((size_t)25 * C2 * C1 * 2);     // conv2 weights bf16
  __bf16* wf1b = (__bf16*)alloc((size_t)NFC1 * FLATK * 2);     // fc1 weights bf16

  for (int e = 0; e < 9; ++e) {
    const float *W1, *B1, *W2, *B2, *Wf1, *Bf1, *Wf2, *Bf2;
    int nv;
    if (e == 0) {
      W1 = gW1; B1 = gb1; W2 = gW2; B2 = gb2;
      Wf1 = gWf1; Bf1 = gbf1; Wf2 = gWf2; Bf2 = gbf2; nv = 8;
    } else {
      int i = e - 1;
      W1  = eW1  + (size_t)i * C1 * 25;        B1  = eb1  + (size_t)i * C1;
      W2  = eW2  + (size_t)i * C2 * C1 * 25;   B2  = eb2  + (size_t)i * C2;
      Wf1 = eWf1 + (size_t)i * NFC1 * FLATK;   Bf1 = ebf1 + (size_t)i * NFC1;
      Wf2 = eWf2 + (size_t)i * 10 * NFC1;      Bf2 = ebf2 + (size_t)i * 10;
      nv = 10;
    }
    prep_w2    <<<(25 * C2 * C1 + 255) / 256, 256, 0, stream>>>(W2, wtap);
    prep_wf1   <<<(NFC1 * FLATK + 255) / 256, 256, 0, stream>>>(Wf1, wf1b);
    conv1_wmma <<<NB, 256, 0, stream>>>(x, W1, B1, y1);
    pool1_kernel<<<(NB * PIX2 * C1 + 255) / 256, 256, 0, stream>>>(y1, pl1);
    conv2_wmma <<<NB, 256, 0, stream>>>(pl1, wtap, B2, y2);
    pool2_kernel<<<(NB * FLATK + 255) / 256, 256, 0, stream>>>(y2, fl2);
    fc1_wmma   <<<32, 256, 0, stream>>>(fl2, wf1b, Bf1, f1o);
    fc2_wmma   <<<16, 256, 0, stream>>>(f1o, Wf2, Bf2, nv, lg + (size_t)e * NB * 16);
  }
  moe_combine<<<(NB + 255) / 256, 256, 0, stream>>>(lg, (float*)d_out);
}
